// Custom_Network_34703335751940
// MI455X (gfx1250) — matrix operand order repaired and checked by reading, not compiled, not run on hardware
//
#include <hip/hip_runtime.h>
#include <hip/hip_bf16.h>

// ---------------- problem constants (fixed by setup_inputs) ----------------
#define SEQ   256
#define BATCH 64
#define HID   1024
#define G4H   4096     // 4*HID
#define NLAY  2
#define NCLS  10
#define SB    (SEQ * BATCH)   // 16384 rows of the bulk GEMM

typedef __attribute__((ext_vector_type(16))) __bf16 v16bf;
typedef __attribute__((ext_vector_type(8)))  float  v8f;

// ---------------- bf16 helpers ----------------
__device__ __forceinline__ unsigned short f2bf(float f) {
    unsigned u = __builtin_bit_cast(unsigned, f);
    u += 0x7FFFu + ((u >> 16) & 1u);           // round-to-nearest-even
    return (unsigned short)(u >> 16);
}
__device__ __forceinline__ float bf2f(unsigned short h) {
    unsigned u = ((unsigned)h) << 16;
    return __builtin_bit_cast(float, u);
}
__device__ __forceinline__ float sigmoidf_(float x) {
    return 1.0f / (1.0f + __expf(-x));
}

// ---------------- WMMA fragment loads (ISA 7.12.2 layouts) ----------------
// A: 16x32 bf16 tile, row-major source with stride ldk (elements).
// lanes 0-15: row=lane,    K chunks {0-7 -> v0..3, 16-23 -> v4..7}
// lanes16-31: row=lane-16, K chunks {8-15,         24-31}
__device__ __forceinline__ v16bf load_a_frag(const unsigned short* __restrict__ base,
                                             int ldk, int lane) {
    int row = lane & 15;
    int kg  = (lane >> 4) << 3;                // 0 or 8
    union { uint4 q[2]; v16bf v; } u;
    u.q[0] = *(const uint4*)(base + row * ldk + kg);
    u.q[1] = *(const uint4*)(base + row * ldk + kg + 16);
    return u.v;
}
// B: 32x16 bf16 tile where B[k][n] = W[r0+n][k], W row-major stride ldk.
// lanes 0-15: col=lane, K {0-7, 16-23} ; lanes 16-31: col=lane-16, K {8-15, 24-31}
__device__ __forceinline__ v16bf load_b_frag(const unsigned short* __restrict__ base,
                                             int ldk, int lane) {
    int col = lane & 15;
    int kg  = (lane >> 4) << 3;                // 0 or 8
    union { uint4 q[2]; v16bf v; } u;
    u.q[0] = *(const uint4*)(base + col * ldk + kg);
    u.q[1] = *(const uint4*)(base + col * ldk + kg + 16);
    return u.v;
}

// ---------------- setup kernels ----------------
__global__ void cvt_f32_bf16(const float* __restrict__ in,
                             unsigned short* __restrict__ out, int n) {
    for (int i = blockIdx.x * blockDim.x + threadIdx.x; i < n;
         i += gridDim.x * blockDim.x)
        out[i] = f2bf(in[i]);
}

__global__ void bias_sum(const float* __restrict__ bias,   // [L][2][4H]
                         float* __restrict__ bsum) {       // [L][4H]
    int i = blockIdx.x * blockDim.x + threadIdx.x;
    if (i >= NLAY * G4H) return;
    int l = i / G4H, r = i - l * G4H;
    bsum[i] = bias[l * 2 * G4H + r] + bias[l * 2 * G4H + G4H + r];
}

__global__ void init_state(unsigned short* __restrict__ hbuf,  // [2][B][H] bf16
                           float* __restrict__ cbuf) {         // [B][H] f32
    int i = blockIdx.x * blockDim.x + threadIdx.x;
    const int n = BATCH * HID;
    if (i >= n) return;
    cbuf[i] = 0.0f;
    hbuf[i] = 0;
    hbuf[n + i] = 0;
}

// ---------------- bulk input-projection GEMM ----------------
// out[M=SB][N=4H] (bf16) = X[M][K=H] (bf16) @ W[N][K] (bf16)^T
// grid: (N/64, M/64), block: 128 threads = 4 waves, each wave a 32x32 reg tile.
__global__ void __launch_bounds__(128)
gemm_xw(const unsigned short* __restrict__ X,
        const unsigned short* __restrict__ W,
        unsigned short* __restrict__ out) {
    const int lane = threadIdx.x & 31;
    const int wav  = threadIdx.x >> 5;              // 0..3
    const int m0   = blockIdx.y * 64 + (wav >> 1) * 32;
    const int n0   = blockIdx.x * 64 + (wav & 1) * 32;

    const v8f zero = {0.f, 0.f, 0.f, 0.f, 0.f, 0.f, 0.f, 0.f};
    v8f acc[2][2];
    acc[0][0] = zero; acc[0][1] = zero; acc[1][0] = zero; acc[1][1] = zero;

    #pragma unroll 2
    for (int k = 0; k < HID; k += 32) {
        v16bf a0 = load_a_frag(X + (size_t)m0 * HID + k, HID, lane);
        v16bf a1 = load_a_frag(X + (size_t)(m0 + 16) * HID + k, HID, lane);
        v16bf b0 = load_b_frag(W + (size_t)n0 * HID + k, HID, lane);
        v16bf b1 = load_b_frag(W + (size_t)(n0 + 16) * HID + k, HID, lane);
        acc[0][0] = __builtin_amdgcn_wmma_f32_16x16x32_bf16(false, a0, false, b0, (short)0, acc[0][0], false, false);
        acc[0][1] = __builtin_amdgcn_wmma_f32_16x16x32_bf16(false, a0, false, b1, (short)0, acc[0][1], false, false);
        acc[1][0] = __builtin_amdgcn_wmma_f32_16x16x32_bf16(false, a1, false, b0, (short)0, acc[1][0], false, false);
        acc[1][1] = __builtin_amdgcn_wmma_f32_16x16x32_bf16(false, a1, false, b1, (short)0, acc[1][1], false, false);
    }

    const int n  = lane & 15;
    const int mh = (lane >> 4) << 3;
    #pragma unroll
    for (int mi = 0; mi < 2; ++mi)
        #pragma unroll
        for (int ni = 0; ni < 2; ++ni)
            #pragma unroll
            for (int v = 0; v < 8; ++v) {
                int row = m0 + mi * 16 + mh + v;
                int col = n0 + ni * 16 + n;
                out[(size_t)row * G4H + col] = f2bf(acc[mi][ni][v]);
            }
}

// ---------------- recurrent LSTM step (K = HID only) ----------------
// grid: 64 blocks (16 hidden units each), 64 threads = 2 waves
// (each wave: 2 batch-tiles x 4 gates = 8 accumulators).
// gates = gate_pre[t] + h_prev @ Whh^T + b ; cell update ; h out (bf16)
__global__ void __launch_bounds__(64)
lstm_step(const unsigned short* __restrict__ gpre_t,  // [B][4H] bf16
          const unsigned short* __restrict__ h_prev,  // [B][H] bf16
          const unsigned short* __restrict__ w_hh,    // [4H][H] bf16
          const float* __restrict__ bsum,             // [4H]
          float* __restrict__ c,                      // [B][H] f32 (in/out)
          unsigned short* __restrict__ h_next,        // [B][H] bf16
          unsigned short* __restrict__ seq_out)       // [B][H] bf16 or null
{
    const int lane = threadIdx.x & 31;
    const int wav  = threadIdx.x >> 5;     // 0..1
    const int m0   = wav << 5;             // batch tiles m0, m0+16
    const int j0   = blockIdx.x << 4;      // hidden-unit base

    const v8f zero = {0.f, 0.f, 0.f, 0.f, 0.f, 0.f, 0.f, 0.f};
    v8f acc[2][4];
    #pragma unroll
    for (int mi = 0; mi < 2; ++mi)
        #pragma unroll
        for (int g = 0; g < 4; ++g) acc[mi][g] = zero;

    #pragma unroll 2
    for (int k = 0; k < HID; k += 32) {
        v16bf a0 = load_a_frag(h_prev + m0 * HID + k, HID, lane);
        v16bf a1 = load_a_frag(h_prev + (m0 + 16) * HID + k, HID, lane);
        #pragma unroll
        for (int g = 0; g < 4; ++g) {
            v16bf b = load_b_frag(w_hh + (size_t)(g * HID + j0) * HID + k, HID, lane);
            acc[0][g] = __builtin_amdgcn_wmma_f32_16x16x32_bf16(
                false, a0, false, b, (short)0, acc[0][g], false, false);
            acc[1][g] = __builtin_amdgcn_wmma_f32_16x16x32_bf16(
                false, a1, false, b, (short)0, acc[1][g], false, false);
        }
    }

    // cell update. acc element v: (m = m0 + 16*mi + v + 8*(lane>=16), n = lane&15)
    const int n  = lane & 15;
    const int j  = j0 + n;
    const int mh = (lane >> 4) << 3;
    const float bi = bsum[0 * HID + j];
    const float bf = bsum[1 * HID + j];
    const float bg = bsum[2 * HID + j];
    const float bo = bsum[3 * HID + j];

    #pragma unroll
    for (int mi = 0; mi < 2; ++mi) {
        #pragma unroll
        for (int v = 0; v < 8; ++v) {
            const int m   = m0 + mi * 16 + mh + v;
            const int idx = m * HID + j;
            const unsigned short* gp = gpre_t + (size_t)m * G4H + j;
            float gi = sigmoidf_(acc[mi][0][v] + bf2f(gp[0 * HID]) + bi);
            float gf = sigmoidf_(acc[mi][1][v] + bf2f(gp[1 * HID]) + bf);
            float gg = tanhf    (acc[mi][2][v] + bf2f(gp[2 * HID]) + bg);
            float go = sigmoidf_(acc[mi][3][v] + bf2f(gp[3 * HID]) + bo);
            float cn = gf * c[idx] + gi * gg;
            c[idx]   = cn;
            unsigned short hb = f2bf(go * tanhf(cn));
            h_next[idx] = hb;
            if (seq_out) seq_out[idx] = hb;
        }
    }
}

// ---------------- final FC: out[b][c] = h.fc_w[c] + fc_b[c] ----------------
__global__ void fc_kernel(const unsigned short* __restrict__ h,  // [B][H] bf16
                          const float* __restrict__ fc_w,        // [C][H]
                          const float* __restrict__ fc_b,        // [C]
                          float* __restrict__ out) {             // [B][C]
    int idx = blockIdx.x * blockDim.x + threadIdx.x;
    if (idx >= BATCH * NCLS) return;
    int b = idx / NCLS, cls = idx - b * NCLS;
    float s = fc_b[cls];
    for (int k = 0; k < HID; ++k)
        s += bf2f(h[b * HID + k]) * fc_w[cls * HID + k];
    out[idx] = s;
}

// ---------------- host launcher ----------------
extern "C" void kernel_launch(void* const* d_in, const int* in_sizes, int n_in,
                              void* d_out, int out_size, void* d_ws, size_t ws_size,
                              hipStream_t stream) {
    const float* x_data = (const float*)d_in[0];  // [S][B][H]
    const float* weight = (const float*)d_in[1];  // [L][2][4H][H]
    const float* bias   = (const float*)d_in[2];  // [L][2][4H]
    const float* fc_w   = (const float*)d_in[3];  // [C][H]
    const float* fc_b   = (const float*)d_in[4];  // [C]
    float* out = (float*)d_out;

    // workspace layout (256B aligned)
    char* ws = (char*)d_ws;
    size_t off = 0;
    auto alloc = [&](size_t bytes) {
        char* p = ws + off;
        off += (bytes + 255) & ~(size_t)255;
        return p;
    };
    unsigned short* wbf  = (unsigned short*)alloc((size_t)NLAY * 2 * G4H * HID * 2);
    float*          bsm  = (float*)         alloc((size_t)NLAY * G4H * 4);
    unsigned short* xbf0 = (unsigned short*)alloc((size_t)SB * HID * 2);
    unsigned short* xbf1 = (unsigned short*)alloc((size_t)SB * HID * 2);
    unsigned short* gpre = (unsigned short*)alloc((size_t)SB * G4H * 2);
    unsigned short* hbuf = (unsigned short*)alloc((size_t)2 * BATCH * HID * 2);
    float*          cbuf = (float*)         alloc((size_t)BATCH * HID * 4);
    (void)ws_size; (void)in_sizes; (void)n_in; (void)out_size;

    // setup: bf16 conversions + bias pre-sum
    {
        int nw = NLAY * 2 * G4H * HID;
        cvt_f32_bf16<<<4096, 256, 0, stream>>>(weight, wbf, nw);
        int nx = SB * HID;
        cvt_f32_bf16<<<4096, 256, 0, stream>>>(x_data, xbf0, nx);
        bias_sum<<<(NLAY * G4H + 255) / 256, 256, 0, stream>>>(bias, bsm);
    }

    const int nhc = BATCH * HID;
    const dim3 ggrid(G4H / 64, SB / 64);            // (64, 256) blocks
    for (int l = 0; l < NLAY; ++l) {
        const unsigned short* xin = (l == 0) ? xbf0 : xbf1;
        unsigned short* seq       = (l == 0) ? xbf1 : nullptr;
        const unsigned short* wih = wbf + (size_t)(l * 2 + 0) * G4H * HID;
        const unsigned short* whh = wbf + (size_t)(l * 2 + 1) * G4H * HID;
        const float* bs           = bsm + (size_t)l * G4H;

        // parallel input projection for all timesteps of this layer
        gemm_xw<<<ggrid, 128, 0, stream>>>(xin, wih, gpre);

        // sequential recurrence
        init_state<<<(nhc + 255) / 256, 256, 0, stream>>>(hbuf, cbuf);
        for (int t = 0; t < SEQ; ++t) {
            lstm_step<<<HID / 16, 64, 0, stream>>>(
                gpre + (size_t)t * BATCH * G4H,
                hbuf + (size_t)(t & 1) * nhc,
                whh, bs, cbuf,
                hbuf + (size_t)((t + 1) & 1) * nhc,
                seq ? (seq + (size_t)t * nhc) : nullptr);
        }
    }

    // after t = SEQ-1, final h is at parity (SEQ & 1) == 0
    fc_kernel<<<(BATCH * NCLS + 255) / 256, 256, 0, stream>>>(hbuf, fc_w, fc_b, out);
}
